// VisionTransformer_63015760167299
// MI455X (gfx1250) — matrix operand order repaired and checked by reading, not compiled, not run on hardware
//
#include <hip/hip_runtime.h>
#include <hip/hip_bf16.h>
#include <math.h>

// ---- CDNA5 WMMA types ----
typedef __attribute__((ext_vector_type(16))) __bf16 v16bf;
typedef __attribute__((ext_vector_type(8)))  __bf16 v8bf;
typedef __attribute__((ext_vector_type(8)))  float  v8f;

#define SP 224   // padded token stride for scores/P/vT (7*32)

__device__ __forceinline__ v8f wmma_bf16(v16bf a, v16bf b, v8f c) {
  return __builtin_amdgcn_wmma_f32_16x16x32_bf16(false, a, false, b, (short)0, c,
                                                 false, false);
}

// Two 16B (8 x bf16) loads -> one 16-element fragment.
__device__ __forceinline__ v16bf load_frag(const __bf16* p0, const __bf16* p1) {
  const v8bf lo = *(const v8bf*)p0;
  const v8bf hi = *(const v8bf*)p1;
  return __builtin_shufflevector(lo, hi, 0,1,2,3,4,5,6,7,8,9,10,11,12,13,14,15);
}
// A fragment (16x32): lane holds K [k0+16*(lane>=16) .. +15], kb = k0 + 8*(lane>=16)
__device__ __forceinline__ v16bf load_a(const __bf16* rowp, int kb) {
  return load_frag(rowp + kb + (kb & 8), rowp + kb + (kb & 8) + 8);
}
// B fragment (32x16) from K-contiguous (transposed) storage:
// lane = out column, holds K [kb..kb+15], kb = k0 + 16*(lane>=16)
__device__ __forceinline__ v16bf load_b(const __bf16* colp, int kb) {
  return load_frag(colp + kb, colp + kb + 8);
}

// =====================================================================
// Weight convert+transpose: W[K][N] f32 -> Wt[N][K] bf16
// =====================================================================
__global__ void wcvt_kernel(const float* __restrict__ W, __bf16* __restrict__ Wt,
                            int K, int N) {
  int idx = blockIdx.x * blockDim.x + threadIdx.x;
  if (idx >= K * N) return;
  int n = idx % N, k = idx / N;
  Wt[(size_t)n * K + k] = (__bf16)W[(size_t)k * N + n];
}

__global__ void zero_bf16_kernel(__bf16* __restrict__ p, int total) {
  int idx = blockIdx.x * blockDim.x + threadIdx.x;
  if (idx < total) p[idx] = (__bf16)0.0f;
}

// =====================================================================
// Generic GEMM, fp32 output: C = A_bf16 @ Bt_bf16^T + bias (+resid)(+gelu)
// One wave per 16x64 tile; per K-step: 2 b128 (A) + 8 b128 (B) + 4 WMMA.
// K % 32 == 0. OOB rows/cols: clamp loads, guard stores.
// =====================================================================
__global__ __launch_bounds__(32)
void gemm_f32out(const __bf16* __restrict__ A, const __bf16* __restrict__ Bt,
                 const float* __restrict__ bias, const float* resid, float* C,
                 int M, int N, int K, int ldc, int do_gelu) {
  const int lane   = threadIdx.x & 31;
  const int row0   = blockIdx.y * 16;
  const int col0   = blockIdx.x * 64;
  const int a_koff = (lane >> 4) << 3;
  const int b_koff = (lane >> 4) << 4;

  int arow = row0 + (lane & 15); if (arow > M - 1) arow = M - 1;
  const __bf16* Arow = A + (size_t)arow * K;
  const __bf16* Bc[4];
#pragma unroll
  for (int nt = 0; nt < 4; ++nt) {
    int bc = col0 + nt * 16 + (lane & 15); if (bc > N - 1) bc = N - 1;
    Bc[nt] = Bt + (size_t)bc * K;
  }

  v8f acc[4];
#pragma unroll
  for (int nt = 0; nt < 4; ++nt) acc[nt] = (v8f)0.0f;

  for (int k0 = 0; k0 < K; k0 += 32) {
    __builtin_prefetch(Arow + k0 + 512, 0, 1);
    const v16bf af = load_a(Arow, k0 + a_koff);
#pragma unroll
    for (int nt = 0; nt < 4; ++nt)
      acc[nt] = wmma_bf16(af, load_b(Bc[nt], k0 + b_koff), acc[nt]);
  }

  const int crow_base = row0 + ((lane >> 4) << 3);
  const int ccol_l    = lane & 15;
#pragma unroll
  for (int nt = 0; nt < 4; ++nt) {
    const int ccol = col0 + nt * 16 + ccol_l;
    if (ccol < N) {
      const float bv = bias ? bias[ccol] : 0.f;
#pragma unroll
      for (int r = 0; r < 8; ++r) {
        const int crow = crow_base + r;
        if (crow < M) {
          float v = acc[nt][r] + bv;
          if (resid)   v += resid[(size_t)crow * ldc + ccol];
          if (do_gelu) v  = 0.5f * v * (1.0f + erff(v * 0.70710678118654752f));
          C[(size_t)crow * ldc + ccol] = v;
        }
      }
    }
  }
}

// =====================================================================
// Generic GEMM, bf16 output (for fc1+GELU -> mlp activations)
// =====================================================================
__global__ __launch_bounds__(32)
void gemm_bf16out(const __bf16* __restrict__ A, const __bf16* __restrict__ Bt,
                  const float* __restrict__ bias, __bf16* __restrict__ C,
                  int M, int N, int K, int ldc, int do_gelu) {
  const int lane   = threadIdx.x & 31;
  const int row0   = blockIdx.y * 16;
  const int col0   = blockIdx.x * 64;
  const int a_koff = (lane >> 4) << 3;
  const int b_koff = (lane >> 4) << 4;

  int arow = row0 + (lane & 15); if (arow > M - 1) arow = M - 1;
  const __bf16* Arow = A + (size_t)arow * K;
  const __bf16* Bc[4];
#pragma unroll
  for (int nt = 0; nt < 4; ++nt) {
    int bc = col0 + nt * 16 + (lane & 15); if (bc > N - 1) bc = N - 1;
    Bc[nt] = Bt + (size_t)bc * K;
  }

  v8f acc[4];
#pragma unroll
  for (int nt = 0; nt < 4; ++nt) acc[nt] = (v8f)0.0f;

  for (int k0 = 0; k0 < K; k0 += 32) {
    __builtin_prefetch(Arow + k0 + 512, 0, 1);
    const v16bf af = load_a(Arow, k0 + a_koff);
#pragma unroll
    for (int nt = 0; nt < 4; ++nt)
      acc[nt] = wmma_bf16(af, load_b(Bc[nt], k0 + b_koff), acc[nt]);
  }

  const int crow_base = row0 + ((lane >> 4) << 3);
  const int ccol_l    = lane & 15;
#pragma unroll
  for (int nt = 0; nt < 4; ++nt) {
    const int ccol = col0 + nt * 16 + ccol_l;
    if (ccol < N) {
      const float bv = bias[ccol];
#pragma unroll
      for (int r = 0; r < 8; ++r) {
        const int crow = crow_base + r;
        if (crow < M) {
          float v = acc[nt][r] + bv;
          if (do_gelu) v = 0.5f * v * (1.0f + erff(v * 0.70710678118654752f));
          C[(size_t)crow * ldc + ccol] = (__bf16)v;
        }
      }
    }
  }
}

// =====================================================================
// QKV GEMM: A=ln_bf16 [3152,768], Bt=qkv_wt [2304,768].
// Epilogue splits: cols [0,1536) -> qkv_bf row-major (Q,K);
// cols [1536,2304) -> vT[b,h,d,token] (token stride SP) for attn*V.
// =====================================================================
__global__ __launch_bounds__(32)
void gemm_qkv(const __bf16* __restrict__ A, const __bf16* __restrict__ Bt,
              const float* __restrict__ bias, __bf16* __restrict__ qkv_bf,
              __bf16* __restrict__ vT) {
  const int lane   = threadIdx.x & 31;
  const int row0   = blockIdx.y * 16;
  const int col0   = blockIdx.x * 64;
  const int a_koff = (lane >> 4) << 3;
  const int b_koff = (lane >> 4) << 4;
  const int K = 768;

  const int arow = row0 + (lane & 15);            // M=3152, always in range
  const __bf16* Arow = A + (size_t)arow * K;
  const __bf16* Bc[4];
#pragma unroll
  for (int nt = 0; nt < 4; ++nt)
    Bc[nt] = Bt + (size_t)(col0 + nt * 16 + (lane & 15)) * K;

  v8f acc[4];
#pragma unroll
  for (int nt = 0; nt < 4; ++nt) acc[nt] = (v8f)0.0f;

  for (int k0 = 0; k0 < K; k0 += 32) {
    const v16bf af = load_a(Arow, k0 + a_koff);
#pragma unroll
    for (int nt = 0; nt < 4; ++nt)
      acc[nt] = wmma_bf16(af, load_b(Bc[nt], k0 + b_koff), acc[nt]);
  }

  const int crow_base = row0 + ((lane >> 4) << 3);
  const int ccol_l    = lane & 15;
#pragma unroll
  for (int nt = 0; nt < 4; ++nt) {
    const int ccol = col0 + nt * 16 + ccol_l;
    const float bv = bias[ccol];
#pragma unroll
    for (int r = 0; r < 8; ++r) {
      const int crow = crow_base + r;
      const float v = acc[nt][r] + bv;
      if (ccol < 1536) {
        qkv_bf[(size_t)crow * 1536 + ccol] = (__bf16)v;
      } else {
        const int h = (ccol - 1536) >> 6, d = (ccol - 1536) & 63;
        const int b = crow / 197, n = crow % 197;
        vT[(((size_t)b * 12 + h) * 64 + d) * SP + n] = (__bf16)v;
      }
    }
  }
}

// =====================================================================
// Attention scores: S[b,h] = (Q @ K^T) * 0.125 (fp32, stride SP).
// Q,K rows are bf16 with contiguous head-dim -> all b128 loads. K=64.
// =====================================================================
__global__ __launch_bounds__(32)
void attn_qk_kernel(const __bf16* __restrict__ qkv_bf, float* __restrict__ scores) {
  const int lane = threadIdx.x & 31;
  const int bh   = blockIdx.z;
  const int b    = bh / 12, h = bh % 12;
  const int row0 = blockIdx.y * 16;
  const int col0 = blockIdx.x * 16;

  int arow = row0 + (lane & 15); if (arow > 196) arow = 196;
  int bcol = col0 + (lane & 15); if (bcol > 196) bcol = 196;

  const __bf16* base = qkv_bf + (size_t)b * 197 * 1536;
  const __bf16* Qrow = base + (size_t)arow * 1536 + h * 64;
  const __bf16* Krow = base + (size_t)bcol * 1536 + 768 + h * 64;
  const int a_koff = (lane >> 4) << 3;
  const int b_koff = (lane >> 4) << 4;

  v8f acc = (v8f)0.0f;
  acc = wmma_bf16(load_a(Qrow, a_koff),      load_b(Krow, b_koff),      acc);
  acc = wmma_bf16(load_a(Qrow, 32 + a_koff), load_b(Krow, 32 + b_koff), acc);

  float* S = scores + (size_t)bh * 197 * SP;
  const int crow_base = row0 + ((lane >> 4) << 3);
  const int ccol      = col0 + (lane & 15);
  if (ccol < 197) {
#pragma unroll
    for (int r = 0; r < 8; ++r) {
      const int crow = crow_base + r;
      if (crow < 197) S[(size_t)crow * SP + ccol] = acc[r] * 0.125f;
    }
  }
}

// =====================================================================
// O = P @ V. P bf16 (stride SP, cols 197..223 exact zero), V transposed
// (vT[bh*64+d][tok], pad tokens zeroed) -> 7 guard-free K-steps of 32.
// One wave per 16x64 tile (full head dim). Output bf16 o[3152,768].
// =====================================================================
__global__ __launch_bounds__(32)
void attn_av_kernel(const __bf16* __restrict__ P, const __bf16* __restrict__ vT,
                    __bf16* __restrict__ o) {
  const int lane = threadIdx.x & 31;
  const int bh   = blockIdx.y;
  const int b    = bh / 12, h = bh % 12;
  const int row0 = blockIdx.x * 16;

  int arow = row0 + (lane & 15); if (arow > 196) arow = 196;
  const __bf16* Prow = P + (size_t)bh * 197 * SP + (size_t)arow * SP;

  const int a_koff = (lane >> 4) << 3;
  const int b_koff = (lane >> 4) << 4;
  const int ccol_l = lane & 15;

  const __bf16* Vc[4];
#pragma unroll
  for (int nt = 0; nt < 4; ++nt)
    Vc[nt] = vT + ((size_t)bh * 64 + nt * 16 + ccol_l) * SP;

  v8f acc[4];
#pragma unroll
  for (int nt = 0; nt < 4; ++nt) acc[nt] = (v8f)0.0f;

#pragma unroll
  for (int k0 = 0; k0 < SP; k0 += 32) {
    const v16bf af = load_a(Prow, k0 + a_koff);
#pragma unroll
    for (int nt = 0; nt < 4; ++nt)
      acc[nt] = wmma_bf16(af, load_b(Vc[nt], k0 + b_koff), acc[nt]);
  }

  const int crow_base = row0 + ((lane >> 4) << 3);
#pragma unroll
  for (int nt = 0; nt < 4; ++nt) {
#pragma unroll
    for (int r = 0; r < 8; ++r) {
      const int crow = crow_base + r;
      if (crow < 197)
        o[((size_t)b * 197 + crow) * 768 + h * 64 + nt * 16 + ccol_l] = (__bf16)acc[nt][r];
    }
  }
}

// =====================================================================
// LayerNorm: fp32 in, bf16 out; one wave32 per 768-wide row.
// =====================================================================
__global__ __launch_bounds__(32)
void layernorm_kernel(const float* __restrict__ x, const float* __restrict__ s,
                      const float* __restrict__ bb, __bf16* __restrict__ y,
                      int in_stride) {
  const int row  = blockIdx.x;
  const int lane = threadIdx.x;
  const float* xr = x + (size_t)row * in_stride;
  float v[24];
  float sum = 0.f;
#pragma unroll
  for (int j = 0; j < 24; ++j) { v[j] = xr[j * 32 + lane]; sum += v[j]; }
#pragma unroll
  for (int m = 16; m >= 1; m >>= 1) sum += __shfl_xor(sum, m, 32);
  const float mean = sum * (1.0f / 768.0f);
  float var = 0.f;
#pragma unroll
  for (int j = 0; j < 24; ++j) { float d = v[j] - mean; var += d * d; }
#pragma unroll
  for (int m = 16; m >= 1; m >>= 1) var += __shfl_xor(var, m, 32);
  const float rstd = rsqrtf(var * (1.0f / 768.0f) + 1e-6f);
  __bf16* yr = y + (size_t)row * 768;
#pragma unroll
  for (int j = 0; j < 24; ++j) {
    const int idx = j * 32 + lane;
    yr[idx] = (__bf16)((v[j] - mean) * rstd * s[idx] + bb[idx]);
  }
}

// =====================================================================
// Softmax: fp32 scores in -> bf16 P out (both stride SP); pad cols get 0.
// =====================================================================
__global__ __launch_bounds__(32)
void softmax_kernel(const float* __restrict__ S, __bf16* __restrict__ P) {
  const int row  = blockIdx.x;            // 192*197 rows
  const int lane = threadIdx.x;
  const float* r = S + (size_t)row * SP;
  float v[7];
  float mx = -1e30f;
#pragma unroll
  for (int j = 0; j < 7; ++j) {
    const int idx = j * 32 + lane;
    const float val = r[idx];
    v[j] = (idx < 197) ? val : -1e30f;
    mx = fmaxf(mx, v[j]);
  }
#pragma unroll
  for (int m = 16; m >= 1; m >>= 1) mx = fmaxf(mx, __shfl_xor(mx, m, 32));
  float sum = 0.f;
#pragma unroll
  for (int j = 0; j < 7; ++j) {
    const int idx = j * 32 + lane;
    v[j] = (idx < 197) ? expf(v[j] - mx) : 0.f;
    sum += v[j];
  }
#pragma unroll
  for (int m = 16; m >= 1; m >>= 1) sum += __shfl_xor(sum, m, 32);
  const float inv = 1.0f / sum;
  __bf16* pr = P + (size_t)row * SP;
#pragma unroll
  for (int j = 0; j < 7; ++j)
    pr[j * 32 + lane] = (__bf16)(v[j] * inv);       // pad cols -> exact 0
}

// =====================================================================
// Patchify: x[16,3,224,224] f32 -> xp[3136,768] bf16, col = c*256+ph*16+pw
// =====================================================================
__global__ void patchify_kernel(const float* __restrict__ x, __bf16* __restrict__ xp,
                                int total) {
  int idx = blockIdx.x * blockDim.x + threadIdx.x;
  if (idx >= total) return;
  int cidx = idx % 768;
  int row  = idx / 768;
  int c  = cidx / 256;
  int r2 = cidx % 256;
  int ph = r2 / 16, pw = r2 % 16;
  int b  = row / 196, p = row % 196;
  int gy = p / 14, gx = p % 14;
  xp[idx] = (__bf16)x[(((size_t)b * 3 + c) * 224 + gy * 16 + ph) * 224 + gx * 16 + pw];
}

// =====================================================================
// Assemble tokens (fp32): t[b,0]=cls+pos[0]; t[b,1+p]=xe[b,p]+pos[1+p]
// =====================================================================
__global__ void assemble_kernel(const float* __restrict__ xe, const float* __restrict__ cls,
                                const float* __restrict__ pos, float* __restrict__ t,
                                int total) {
  int idx = blockIdx.x * blockDim.x + threadIdx.x;
  if (idx >= total) return;
  int d   = idx % 768;
  int rem = idx / 768;
  int n   = rem % 197;
  int b   = rem / 197;
  float v = (n == 0) ? cls[d] : xe[((size_t)b * 196 + (n - 1)) * 768 + d];
  t[idx] = v + pos[n * 768 + d];
}

// =====================================================================
// Host-side orchestration
// =====================================================================
extern "C" void kernel_launch(void* const* d_in, const int* in_sizes, int n_in,
                              void* d_out, int out_size, void* d_ws, size_t ws_size,
                              hipStream_t stream) {
  (void)in_sizes; (void)n_in; (void)out_size; (void)ws_size;
  const float* x        = (const float*)d_in[0];
  const float* patch_w  = (const float*)d_in[1];
  const float* patch_b  = (const float*)d_in[2];
  const float* cls_tok  = (const float*)d_in[3];
  const float* pos_emb  = (const float*)d_in[4];
  const float* ln1_s    = (const float*)d_in[5];
  const float* ln1_b    = (const float*)d_in[6];
  const float* qkv_w    = (const float*)d_in[7];
  const float* qkv_b    = (const float*)d_in[8];
  const float* proj_w   = (const float*)d_in[9];
  const float* proj_b   = (const float*)d_in[10];
  const float* ln2_s    = (const float*)d_in[11];
  const float* ln2_b    = (const float*)d_in[12];
  const float* fc1_w    = (const float*)d_in[13];
  const float* fc1_b    = (const float*)d_in[14];
  const float* fc2_w    = (const float*)d_in[15];
  const float* fc2_b    = (const float*)d_in[16];
  const float* lnf_s    = (const float*)d_in[17];
  const float* lnf_b    = (const float*)d_in[18];
  const float* head_w   = (const float*)d_in[19];
  const float* head_b   = (const float*)d_in[20];
  float* out = (float*)d_out;

  const int ROWS = 16 * 197;   // 3152

  // ---- workspace carve-up ----
  char* w = (char*)d_ws;
  float* t   = (float*)w;  w += (size_t)ROWS * 768 * 4;          // residual stream
  float* sc  = (float*)w;  w += (size_t)192 * 197 * SP * 4;      // fp32 scores
  float* xe  = sc;                                               // alias (patch GEMM out)
  __bf16* Pb   = (__bf16*)w; w += (size_t)192 * 197 * SP * 2;    // softmax out
  __bf16* qkvb = (__bf16*)w; w += (size_t)ROWS * 1536 * 2;       // Q,K rows
  __bf16* vT   = (__bf16*)w; w += (size_t)192 * 64 * SP * 2;     // V transposed
  __bf16* lnb  = (__bf16*)w; w += (size_t)ROWS * 768 * 2;        // LN out
  __bf16* ob   = (__bf16*)w; w += (size_t)ROWS * 768 * 2;        // attn out
  __bf16* mlpb = (__bf16*)w; w += (size_t)ROWS * 3072 * 2;       // fc1 out
  __bf16* xpb    = mlpb;                                         // alias (patchify out)
  __bf16* clslnb = lnb;                                          // alias (final LN)
  // bf16, transposed weights
  __bf16* patch_wt = (__bf16*)w; w += (size_t)768 * 768 * 2;
  __bf16* qkv_wt   = (__bf16*)w; w += (size_t)12 * 2304 * 768 * 2;
  __bf16* proj_wt  = (__bf16*)w; w += (size_t)12 * 768 * 768 * 2;
  __bf16* fc1_wt   = (__bf16*)w; w += (size_t)12 * 3072 * 768 * 2;
  __bf16* fc2_wt   = (__bf16*)w; w += (size_t)12 * 768 * 3072 * 2;
  __bf16* head_wt  = (__bf16*)w; w += (size_t)1000 * 768 * 2;

  const dim3 b32(32, 1, 1);

  // ---- convert + transpose all weights to bf16 [N][K] ----
  wcvt_kernel<<<(768 * 768 + 255) / 256, 256, 0, stream>>>(patch_w, patch_wt, 768, 768);
  wcvt_kernel<<<(768 * 1000 + 255) / 256, 256, 0, stream>>>(head_w, head_wt, 768, 1000);
  for (int l = 0; l < 12; ++l) {
    wcvt_kernel<<<(768 * 2304 + 255) / 256, 256, 0, stream>>>(
        qkv_w + (size_t)l * 768 * 2304, qkv_wt + (size_t)l * 2304 * 768, 768, 2304);
    wcvt_kernel<<<(768 * 768 + 255) / 256, 256, 0, stream>>>(
        proj_w + (size_t)l * 768 * 768, proj_wt + (size_t)l * 768 * 768, 768, 768);
    wcvt_kernel<<<(768 * 3072 + 255) / 256, 256, 0, stream>>>(
        fc1_w + (size_t)l * 768 * 3072, fc1_wt + (size_t)l * 3072 * 768, 768, 3072);
    wcvt_kernel<<<(3072 * 768 + 255) / 256, 256, 0, stream>>>(
        fc2_w + (size_t)l * 3072 * 768, fc2_wt + (size_t)l * 768 * 3072, 3072, 768);
  }
  // zero vT once (pad tokens must read as 0; valid region rewritten each layer)
  zero_bf16_kernel<<<(192 * 64 * SP + 255) / 256, 256, 0, stream>>>(vT, 192 * 64 * SP);

  // ---- patch embed ----
  {
    int total = 3136 * 768;
    patchify_kernel<<<(total + 255) / 256, 256, 0, stream>>>(x, xpb, total);
    gemm_f32out<<<dim3(12, 196), b32, 0, stream>>>(
        xpb, patch_wt, patch_b, nullptr, xe, 3136, 768, 768, 768, 0);
    total = ROWS * 768;
    assemble_kernel<<<(total + 255) / 256, 256, 0, stream>>>(xe, cls_tok, pos_emb, t, total);
  }

  // ---- transformer blocks ----
  for (int l = 0; l < 12; ++l) {
    const float* l1s = ln1_s + (size_t)l * 768;
    const float* l1b = ln1_b + (size_t)l * 768;
    const float* qb  = qkv_b + (size_t)l * 2304;
    const float* pb  = proj_b + (size_t)l * 768;
    const float* l2s = ln2_s + (size_t)l * 768;
    const float* l2b = ln2_b + (size_t)l * 768;
    const float* f1b = fc1_b + (size_t)l * 3072;
    const float* f2b = fc2_b + (size_t)l * 768;
    const __bf16* qw  = qkv_wt  + (size_t)l * 2304 * 768;
    const __bf16* pw  = proj_wt + (size_t)l * 768 * 768;
    const __bf16* f1w = fc1_wt  + (size_t)l * 3072 * 768;
    const __bf16* f2w = fc2_wt  + (size_t)l * 768 * 3072;

    layernorm_kernel<<<ROWS, b32, 0, stream>>>(t, l1s, l1b, lnb, 768);

    gemm_qkv<<<dim3(36, 197), b32, 0, stream>>>(lnb, qw, qb, qkvb, vT);

    attn_qk_kernel<<<dim3(13, 13, 192), b32, 0, stream>>>(qkvb, sc);
    softmax_kernel<<<192 * 197, b32, 0, stream>>>(sc, Pb);
    attn_av_kernel<<<dim3(13, 192), b32, 0, stream>>>(Pb, vT, ob);

    gemm_f32out<<<dim3(12, 197), b32, 0, stream>>>(
        ob, pw, pb, t, t, ROWS, 768, 768, 768, 0);

    layernorm_kernel<<<ROWS, b32, 0, stream>>>(t, l2s, l2b, lnb, 768);

    gemm_bf16out<<<dim3(48, 197), b32, 0, stream>>>(
        lnb, f1w, f1b, mlpb, ROWS, 3072, 768, 3072, 1);

    gemm_f32out<<<dim3(12, 197), b32, 0, stream>>>(
        mlpb, f2w, f2b, t, t, ROWS, 768, 3072, 768, 0);
  }

  // ---- head: LN on cls rows, then [16,768]@[768,1000] ----
  layernorm_kernel<<<16, b32, 0, stream>>>(t, lnf_s, lnf_b, clslnb, 197 * 768);
  gemm_f32out<<<dim3(16, 1), b32, 0, stream>>>(
      clslnb, head_wt, head_b, nullptr, out, 16, 1000, 768, 1000, 0);
}
